// Decoder_30983894073702
// MI455X (gfx1250) — matrix operand order repaired and checked by reading, not compiled, not run on hardware
//
#include <hip/hip_runtime.h>

// ---------------- Constants (match reference) ----------------
#define LATENT 512
#define HIDDEN 1024
#define OUT_DIM 256
#define INPUT_DIM 256
#define BATCH 128
#define SEQLEN 512
#define BH (BATCH * HIDDEN)

typedef __attribute__((ext_vector_type(16))) __bf16 v16bf;
typedef __attribute__((ext_vector_type(8)))  __bf16 v8bf;
typedef __attribute__((ext_vector_type(8)))  float  v8f;

__device__ __forceinline__ unsigned short f2bf(float f) {
  unsigned int u = __float_as_uint(f);
  u += 0x7FFFu + ((u >> 16) & 1u);   // round-to-nearest-even
  return (unsigned short)(u >> 16);
}
__device__ __forceinline__ float sigf(float x) {
  return 1.0f / (1.0f + __expf(-x));
}

// ---------------- GEMM parameter block ----------------
// C[M,N] = A[M,K] (bf16, row-major, ld=lda) x W[N,K]^T (bf16, row-major, ld=ldw)
struct GemmP {
  const unsigned short* A; int lda;
  const unsigned short* W; int ldw;
  const float* bias;            // per-column bias (may be null)
  float* Cf; int ldc;           // fp32 dst (STORE_F32 / GATE_C)
  unsigned short* Cbf;          // bf16 dst (FC1), ld = N
  const float* st;              // gate modes: pre-blend state [r*H+n] fp32
  const float* lat;             // gate modes: latent state   [b*H+n] fp32
  const float* glat;            // gate modes: latent gate pre-activation [b*H+n]
  unsigned short* dst0; int dld0; int doff0;  // GATE_H: layer-0 bf16 dst slot
  unsigned short* dst1; int dld1; int doff1;  // GATE_H: layer-1 bf16 dst slot
  int M, N, K;
};

// Each wave: 4 M-tiles x 2 N-tiles (64x32 patch), 8 fp32 accumulators.
// Block = 8 waves as 2(M) x 4(N) -> 128x128 patch. Grid = (N/128, M/128).
// Per K-step: 8 A-loads + 4 B-loads feed 8 WMMAs (1.5 loads/WMMA).
// MODE: 0 = store fp32 (+bias), 1 = relu(+bias) -> bf16,
//       3 = sigmoid gate-blend of h -> bf16 into next-step A slots,
//       4 = sigmoid gate-blend of c -> fp32 in place.
template <int MODE>
__global__ __launch_bounds__(256) void gemm_wmma(GemmP p) {
  const int wave = threadIdx.x >> 5;
  const int lane = threadIdx.x & 31;
  const int row  = lane & 15;
  const int half = lane >> 4;
  const int wm = wave & 1;                       // 2 waves along M
  const int wn = wave >> 1;                      // 4 waves along N
  const int mbase = blockIdx.y * 128 + wm * 64;  // 4 M-tiles per wave
  const int nbase = blockIdx.x * 128 + wn * 32;  // 2 N-tiles per wave

  // B fragment base: lane(row,half) -> W row, 16 contiguous K at half*16
  const unsigned short* Wp = p.W + (size_t)(nbase + row) * p.ldw + half * 16;
  // A fragment base: lane(row,half) -> 16 contiguous K at half*16
  const unsigned short* Ap = p.A + (size_t)(mbase + row) * p.lda + half * 16;
  const size_t arow = (size_t)16 * p.lda;
  const size_t wrow = (size_t)16 * p.ldw;

  v8f acc[4][2];
#pragma unroll
  for (int m = 0; m < 4; ++m)
#pragma unroll
    for (int j = 0; j < 2; ++j) acc[m][j] = (v8f){};

  for (int k = 0; k < p.K; k += 32) {
    v16bf b[2];
#pragma unroll
    for (int j = 0; j < 2; ++j)
      b[j] = *(const v16bf*)(Wp + (size_t)j * wrow + k);
    v16bf a[4];
#pragma unroll
    for (int m = 0; m < 4; ++m) {
      const unsigned short* ap = Ap + (size_t)m * arow + k;
      v8bf alo = *(const v8bf*)(ap);
      v8bf ahi = *(const v8bf*)(ap + 8);
      a[m] = __builtin_shufflevector(alo, ahi,
               0,1,2,3,4,5,6,7,8,9,10,11,12,13,14,15);
    }
#pragma unroll
    for (int m = 0; m < 4; ++m)
#pragma unroll
      for (int j = 0; j < 2; ++j)
        acc[m][j] = __builtin_amdgcn_wmma_f32_16x16x32_bf16(
            false, a[m], false, b[j], (short)0, acc[m][j], false, false);
  }

#pragma unroll
  for (int j = 0; j < 2; ++j) {
    const int n = nbase + j * 16 + row;           // D layout: lane&15 = column
    const float bias = p.bias ? p.bias[n] : 0.0f;
#pragma unroll
    for (int m = 0; m < 4; ++m) {
#pragma unroll
      for (int v = 0; v < 8; ++v) {
        const int r = mbase + m * 16 + v + half * 8;  // D: VGPR v = row (+8 hi)
        float x = acc[m][j][v];
        if (MODE == 0) {
          p.Cf[(size_t)r * p.ldc + n] = x + bias;
        } else if (MODE == 1) {
          float y = x + bias;
          p.Cbf[(size_t)r * p.N + n] = f2bf(y > 0.0f ? y : 0.0f);
        } else if (MODE == 3) {
          const int b = r & (BATCH - 1);
          float g = sigf(x + p.glat[(size_t)b * HIDDEN + n]);
          float hb = g * p.st[(size_t)r * HIDDEN + n] +
                     (1.0f - g) * p.lat[(size_t)b * HIDDEN + n];
          unsigned short hv = f2bf(hb);
          if (r < BATCH) p.dst0[(size_t)r * p.dld0 + p.doff0 + n] = hv;
          else           p.dst1[(size_t)(r - BATCH) * p.dld1 + p.doff1 + n] = hv;
        } else {  // MODE == 4
          const int b = r & (BATCH - 1);
          float g = sigf(x + p.glat[(size_t)b * HIDDEN + n]);
          float cb = g * p.st[(size_t)r * HIDDEN + n] +
                     (1.0f - g) * p.lat[(size_t)b * HIDDEN + n];
          p.Cf[(size_t)r * HIDDEN + n] = cb;  // in-place blend of c-state
        }
      }
    }
  }
}

// ---------------- LSTM cell nonlinearity ----------------
__global__ void lstm_cell(const float* __restrict__ G, const float* __restrict__ bsum,
                          float* cstate_l, float* hstate_l,
                          unsigned short* hn_bf_l, unsigned short* cn_bf_l,
                          unsigned short* a_next, int a_ld, int a_off) {
  int idx = blockIdx.x * blockDim.x + threadIdx.x;
  if (idx >= BH) return;
  int b = idx >> 10, j = idx & (HIDDEN - 1);
  const float* g = G + (size_t)b * (4 * HIDDEN);
  float ig = sigf(g[j]              + bsum[j]);
  float fg = sigf(g[HIDDEN + j]     + bsum[HIDDEN + j]);
  float gg = tanhf(g[2 * HIDDEN + j] + bsum[2 * HIDDEN + j]);
  float og = sigf(g[3 * HIDDEN + j] + bsum[3 * HIDDEN + j]);
  float c = fg * cstate_l[idx] + ig * gg;
  float h = og * tanhf(c);
  cstate_l[idx] = c;              // pre-blend c_new (gate_c blends in place)
  hstate_l[idx] = h;
  hn_bf_l[idx] = f2bf(h);
  cn_bf_l[idx] = f2bf(c);
  if (a_next) a_next[(size_t)b * a_ld + a_off + j] = f2bf(h);  // feed layer-1 A
}

// ---------------- prep helpers ----------------
__global__ void cvt_f32_bf16(unsigned short* dst, int dld, const float* src,
                             int sld, int scol, int rows, int cols) {
  long i = (long)blockIdx.x * blockDim.x + threadIdx.x;
  if (i >= (long)rows * cols) return;
  int r = (int)(i / cols), c = (int)(i % cols);
  dst[(size_t)r * dld + c] = f2bf(src[(size_t)r * sld + scol + c]);
}

__global__ void vec_add(float* dst, const float* a, const float* b, int n) {
  int i = blockIdx.x * blockDim.x + threadIdx.x;
  if (i < n) dst[i] = a[i] + b[i];
}

// tanh finish of latent pre-activation + scatter copies
__global__ void latent_finish(float* pre, unsigned short* bfcopy,
                              unsigned short* d0, int ld0, int off0,
                              unsigned short* d1, int ld1, int off1,
                              float* s0, float* s1) {
  int i = blockIdx.x * blockDim.x + threadIdx.x;
  if (i >= BH) return;
  float t = tanhf(pre[i]);
  pre[i] = t;
  int b = i >> 10, j = i & (HIDDEN - 1);
  unsigned short tv = f2bf(t);
  if (bfcopy) bfcopy[i] = tv;
  if (d0) d0[(size_t)b * ld0 + off0 + j] = tv;
  if (d1) d1[(size_t)b * ld1 + off1 + j] = tv;
  if (s0) s0[i] = t;
  if (s1) s1[i] = t;
}

__global__ void init_x0(unsigned short* A0, const float* start) {
  int i = blockIdx.x * blockDim.x + threadIdx.x;
  if (i >= BATCH * INPUT_DIM) return;
  int b = i >> 8, j = i & (INPUT_DIM - 1);
  A0[(size_t)b * (INPUT_DIM + HIDDEN) + j] = f2bf(start[j]);
}

// ---------------- host ----------------
extern "C" void kernel_launch(void* const* d_in, const int* in_sizes, int n_in,
                              void* d_out, int out_size, void* d_ws, size_t ws_size,
                              hipStream_t stream) {
  (void)in_sizes; (void)n_in; (void)out_size; (void)ws_size;
  const float* z      = (const float*)d_in[0];
  const float* start  = (const float*)d_in[2];
  const float* W_ih0  = (const float*)d_in[3];
  const float* W_hh0  = (const float*)d_in[4];
  const float* b_ih0  = (const float*)d_in[5];
  const float* b_hh0  = (const float*)d_in[6];
  const float* W_ih1  = (const float*)d_in[7];
  const float* W_hh1  = (const float*)d_in[8];
  const float* b_ih1  = (const float*)d_in[9];
  const float* b_hh1  = (const float*)d_in[10];
  const float* W_lh   = (const float*)d_in[11];
  const float* b_lh   = (const float*)d_in[12];
  const float* W_lc   = (const float*)d_in[13];
  const float* b_lc   = (const float*)d_in[14];
  const float* W_fc1  = (const float*)d_in[15];
  const float* b_fc1  = (const float*)d_in[16];
  const float* W_fc2  = (const float*)d_in[17];
  const float* b_fc2  = (const float*)d_in[18];
  const float* Wg_h   = (const float*)d_in[19];
  const float* bg_h   = (const float*)d_in[20];
  const float* Wg_c   = (const float*)d_in[21];
  const float* bg_c   = (const float*)d_in[22];
  float* out = (float*)d_out;

  // Workspace layout
  char* wp = (char*)d_ws;
  auto alloc = [&](size_t bytes) -> void* {
    void* p = wp; wp += (bytes + 255) & ~(size_t)255; return p;
  };
  const int K0 = INPUT_DIM + HIDDEN;   // 1280
  const int K1 = 2 * HIDDEN;           // 2048
  unsigned short* W0cat = (unsigned short*)alloc((size_t)4 * HIDDEN * K0 * 2);
  unsigned short* W1cat = (unsigned short*)alloc((size_t)4 * HIDDEN * K1 * 2);
  unsigned short* Wfc1b = (unsigned short*)alloc((size_t)HIDDEN * HIDDEN * 2);
  unsigned short* Wfc2b = (unsigned short*)alloc((size_t)OUT_DIM * HIDDEN * 2);
  unsigned short* WgHL  = (unsigned short*)alloc((size_t)HIDDEN * HIDDEN * 2);
  unsigned short* WgHR  = (unsigned short*)alloc((size_t)HIDDEN * HIDDEN * 2);
  unsigned short* WgCL  = (unsigned short*)alloc((size_t)HIDDEN * HIDDEN * 2);
  unsigned short* WgCR  = (unsigned short*)alloc((size_t)HIDDEN * HIDDEN * 2);
  unsigned short* Wlhb  = (unsigned short*)alloc((size_t)HIDDEN * LATENT * 2);
  unsigned short* Wlcb  = (unsigned short*)alloc((size_t)HIDDEN * LATENT * 2);
  unsigned short* zb    = (unsigned short*)alloc((size_t)BATCH * LATENT * 2);
  unsigned short* A0    = (unsigned short*)alloc((size_t)BATCH * K0 * 2);
  unsigned short* A1    = (unsigned short*)alloc((size_t)BATCH * K1 * 2);
  unsigned short* Hnbf  = (unsigned short*)alloc((size_t)2 * BH * 2);
  unsigned short* Cnbf  = (unsigned short*)alloc((size_t)2 * BH * 2);
  unsigned short* Tbuf  = (unsigned short*)alloc((size_t)BH * 2);
  unsigned short* hzbf  = (unsigned short*)alloc((size_t)BH * 2);
  unsigned short* czbf  = (unsigned short*)alloc((size_t)BH * 2);
  float* G0     = (float*)alloc((size_t)BATCH * 4 * HIDDEN * 4);
  float* G1     = (float*)alloc((size_t)BATCH * 4 * HIDDEN * 4);
  float* hstate = (float*)alloc((size_t)2 * BH * 4);
  float* cstate = (float*)alloc((size_t)2 * BH * 4);
  float* h_z    = (float*)alloc((size_t)BH * 4);
  float* c_z    = (float*)alloc((size_t)BH * 4);
  float* GhLat  = (float*)alloc((size_t)BH * 4);
  float* GcLat  = (float*)alloc((size_t)BH * 4);
  float* b0sum  = (float*)alloc((size_t)4 * HIDDEN * 4);
  float* b1sum  = (float*)alloc((size_t)4 * HIDDEN * 4);

  auto cvt = [&](unsigned short* dst, int dld, const float* src, int sld,
                 int scol, int rows, int cols) {
    long n = (long)rows * cols;
    cvt_f32_bf16<<<(unsigned)((n + 255) / 256), 256, 0, stream>>>(
        dst, dld, src, sld, scol, rows, cols);
  };
  auto gemm = [&](int mode, const unsigned short* A, int lda,
                  const unsigned short* W, int ldw, const float* bias,
                  float* Cf, int ldc, unsigned short* Cbf,
                  const float* st, const float* lat, const float* glat,
                  unsigned short* dst0, int dld0, int doff0,
                  unsigned short* dst1, int dld1, int doff1,
                  int M, int N, int K) {
    GemmP p;
    p.A = A; p.lda = lda; p.W = W; p.ldw = ldw; p.bias = bias;
    p.Cf = Cf; p.ldc = ldc; p.Cbf = Cbf;
    p.st = st; p.lat = lat; p.glat = glat;
    p.dst0 = dst0; p.dld0 = dld0; p.doff0 = doff0;
    p.dst1 = dst1; p.dld1 = dld1; p.doff1 = doff1;
    p.M = M; p.N = N; p.K = K;
    dim3 grid((unsigned)(N / 128), (unsigned)(M / 128));
    switch (mode) {
      case 0: gemm_wmma<0><<<grid, 256, 0, stream>>>(p); break;
      case 1: gemm_wmma<1><<<grid, 256, 0, stream>>>(p); break;
      case 3: gemm_wmma<3><<<grid, 256, 0, stream>>>(p); break;
      default: gemm_wmma<4><<<grid, 256, 0, stream>>>(p); break;
    }
  };

  // ---- one-time prep: weight conversion (bf16) + concatenated layouts ----
  cvt(W0cat, K0, W_ih0, INPUT_DIM, 0, 4 * HIDDEN, INPUT_DIM);
  cvt(W0cat + INPUT_DIM, K0, W_hh0, HIDDEN, 0, 4 * HIDDEN, HIDDEN);
  cvt(W1cat, K1, W_ih1, HIDDEN, 0, 4 * HIDDEN, HIDDEN);
  cvt(W1cat + HIDDEN, K1, W_hh1, HIDDEN, 0, 4 * HIDDEN, HIDDEN);
  cvt(Wfc1b, HIDDEN, W_fc1, HIDDEN, 0, HIDDEN, HIDDEN);
  cvt(Wfc2b, HIDDEN, W_fc2, HIDDEN, 0, OUT_DIM, HIDDEN);
  cvt(WgHL, HIDDEN, Wg_h, 2 * HIDDEN, 0, HIDDEN, HIDDEN);
  cvt(WgHR, HIDDEN, Wg_h, 2 * HIDDEN, HIDDEN, HIDDEN, HIDDEN);
  cvt(WgCL, HIDDEN, Wg_c, 2 * HIDDEN, 0, HIDDEN, HIDDEN);
  cvt(WgCR, HIDDEN, Wg_c, 2 * HIDDEN, HIDDEN, HIDDEN, HIDDEN);
  cvt(Wlhb, LATENT, W_lh, LATENT, 0, HIDDEN, LATENT);
  cvt(Wlcb, LATENT, W_lc, LATENT, 0, HIDDEN, LATENT);
  cvt(zb, LATENT, z, LATENT, 0, BATCH, LATENT);
  vec_add<<<16, 256, 0, stream>>>(b0sum, b_ih0, b_hh0, 4 * HIDDEN);
  vec_add<<<16, 256, 0, stream>>>(b1sum, b_ih1, b_hh1, 4 * HIDDEN);

  // ---- latent -> initial states (WMMA + tanh) ----
  gemm(0, zb, LATENT, Wlhb, LATENT, b_lh, h_z, HIDDEN, nullptr,
       nullptr, nullptr, nullptr, nullptr, 0, 0, nullptr, 0, 0,
       BATCH, HIDDEN, LATENT);
  latent_finish<<<512, 256, 0, stream>>>(h_z, hzbf,
      A0, K0, INPUT_DIM, A1, K1, HIDDEN, nullptr, nullptr);
  gemm(0, zb, LATENT, Wlcb, LATENT, b_lc, c_z, HIDDEN, nullptr,
       nullptr, nullptr, nullptr, nullptr, 0, 0, nullptr, 0, 0,
       BATCH, HIDDEN, LATENT);
  latent_finish<<<512, 256, 0, stream>>>(c_z, czbf,
      nullptr, 0, 0, nullptr, 0, 0, cstate, cstate + BH);

  // constant latent halves of the overwrite gates (+bias folded in)
  gemm(0, hzbf, HIDDEN, WgHR, HIDDEN, bg_h, GhLat, HIDDEN, nullptr,
       nullptr, nullptr, nullptr, nullptr, 0, 0, nullptr, 0, 0,
       BATCH, HIDDEN, HIDDEN);
  gemm(0, czbf, HIDDEN, WgCR, HIDDEN, bg_c, GcLat, HIDDEN, nullptr,
       nullptr, nullptr, nullptr, nullptr, 0, 0, nullptr, 0, 0,
       BATCH, HIDDEN, HIDDEN);
  init_x0<<<128, 256, 0, stream>>>(A0, start);

  // ---- 512 recurrent steps, 8 kernels each ----
  for (int t = 0; t < SEQLEN; ++t) {
    // layer 0: gates + cell
    gemm(0, A0, K0, W0cat, K0, nullptr, G0, 4 * HIDDEN, nullptr,
         nullptr, nullptr, nullptr, nullptr, 0, 0, nullptr, 0, 0,
         BATCH, 4 * HIDDEN, K0);
    lstm_cell<<<512, 256, 0, stream>>>(G0, b0sum, cstate, hstate,
        Hnbf, Cnbf, A1, K1, 0);
    // layer 1: gates + cell
    gemm(0, A1, K1, W1cat, K1, nullptr, G1, 4 * HIDDEN, nullptr,
         nullptr, nullptr, nullptr, nullptr, 0, 0, nullptr, 0, 0,
         BATCH, 4 * HIDDEN, K1);
    lstm_cell<<<512, 256, 0, stream>>>(G1, b1sum, cstate + BH, hstate + BH,
        Hnbf + BH, Cnbf + BH, nullptr, 0, 0);
    // head: relu(h1 @ Wfc1^T + b) @ Wfc2^T + b -> out[:, t, :]
    gemm(1, Hnbf + BH, HIDDEN, Wfc1b, HIDDEN, b_fc1, nullptr, 0, Tbuf,
         nullptr, nullptr, nullptr, nullptr, 0, 0, nullptr, 0, 0,
         BATCH, HIDDEN, HIDDEN);
    gemm(0, Tbuf, HIDDEN, Wfc2b, HIDDEN, b_fc2,
         out + (size_t)t * OUT_DIM, SEQLEN * OUT_DIM, nullptr,
         nullptr, nullptr, nullptr, nullptr, 0, 0, nullptr, 0, 0,
         BATCH, OUT_DIM, HIDDEN);
    // memory overwrite: blended h -> next-step A slots; blended c in place
    gemm(3, Hnbf, HIDDEN, WgHL, HIDDEN, nullptr, nullptr, 0, nullptr,
         hstate, h_z, GhLat, A0, K0, INPUT_DIM, A1, K1, HIDDEN,
         2 * BATCH, HIDDEN, HIDDEN);
    gemm(4, Cnbf, HIDDEN, WgCL, HIDDEN, nullptr, cstate, HIDDEN, nullptr,
         cstate, c_z, GcLat, nullptr, 0, 0, nullptr, 0, 0,
         2 * BATCH, HIDDEN, HIDDEN);
  }
}
